// PredictionIntraAttention_90744069030180
// MI455X (gfx1250) — matrix operand order repaired and checked by reading, not compiled, not run on hardware
//
#include <hip/hip_runtime.h>
#include <hip/hip_bf16.h>

typedef __attribute__((ext_vector_type(16))) _Float16 v16h;
typedef __attribute__((ext_vector_type(8)))  _Float16 v8h;
typedef __attribute__((ext_vector_type(8)))  float    v8f;

#define B_ 1024
#define C_ 32
#define T_ 9
#define L_ 24
#define H_ 8
#define D_ 4
#define MPAD 224   // conv GEMM M rounded to 14 tiles
#define TP 12      // padded T rows in conv LDS tile

// ---- WMMA 16x16x32 f16 element maps (ISA 7.12.2, wave32) ----
__device__ __forceinline__ int wmma_a_k(int lane, int e) {
    return (e & 7) + ((e >> 3) << 4) + ((lane >> 4) << 3);
}
__device__ __forceinline__ int wmma_b_k(int lane, int e) {
    return (e & 7) + ((e >> 3) << 4) + ((lane >> 4) << 3);
}

// =====================================================================
// Prep kernel: pre-swizzle all GEMM weights into lane-ordered f16 WMMA
// B-fragments (32 contiguous bytes per lane per fragment). Runs once.
//  fragA: 8 frags  = {Wq,Wk,Wv,Wm} x {nt0,nt1}
//  fragC: 18 frags = {nt0,nt1} x {9 conv taps}
// =====================================================================
__global__ __launch_bounds__(32)
void prep_frag_kernel(const float* __restrict__ Wq, const float* __restrict__ Wk,
                      const float* __restrict__ Wv, const float* __restrict__ Wm,
                      const float* __restrict__ convw,
                      _Float16* __restrict__ fragA, _Float16* __restrict__ fragC)
{
    const int f    = blockIdx.x;     // 0..25
    const int lane = threadIdx.x;
    if (f < 8) {
        const int mat = f >> 1, nt = f & 1;
        const float* W = (mat == 0) ? Wq : (mat == 1) ? Wk : (mat == 2) ? Wv : Wm;
        const int n = (lane & 15) + nt * 16;
#pragma unroll
        for (int e = 0; e < 16; ++e)
            fragA[(f * 32 + lane) * 16 + e] = (_Float16)W[wmma_b_k(lane, e) * C_ + n];
    } else {
        const int g  = f - 8;            // 0..17
        const int nt = g / 9, tap = g % 9;
        const int kh = tap / 3, kw = tap % 3;
        const int co = (lane & 15) + nt * 16;
#pragma unroll
        for (int e = 0; e < 16; ++e)
            fragC[(g * 32 + lane) * 16 + e] =
                (_Float16)convw[((co * C_ + wmma_b_k(lane, e)) * 3 + kh) * 3 + kw];
    }
}

// =====================================================================
// Kernel A: per-(b,t) fused QKV proj + intra attention + out proj.
// One wave per block. WMMA everywhere a GEMM hides:
//   - QKV / out projections (32x32, K=32)
//   - head-mixed logits:  (q * scale_H2) @ k^T, 8 x (24x32x24)   [32 wmma]
//   - ctx apply:          att_flat(24x192) @ Bctx(192x32)        [24 wmma]
// Only l2norm + softmax remain VALU (inherently scalar).
// =====================================================================
__global__ __launch_bounds__(32)
void attn_kernel(const float* __restrict__ x,
                 const float* __restrict__ bq, const float* __restrict__ bk,
                 const float* __restrict__ bv, const float* __restrict__ bm,
                 const float* __restrict__ Wlogit, const float* __restrict__ Wctx,
                 const float* __restrict__ relb,
                 const v16h* __restrict__ fragA,
                 float* __restrict__ xatt)
{
    const int bt   = blockIdx.x;
    const int b    = bt / T_;
    const int t    = bt % T_;
    const int lane = threadIdx.x;
    const int lo   = lane & 15;
    const int hi   = lane >> 4;

    __shared__ _Float16 xs[32][C_];         // xp tile fp16, rows 24-31 zero
    __shared__ float    qf[32][C_], kf[32][C_], vf[32][C_];
    __shared__ _Float16 qh[32][C_], kh[32][C_], vh[32][C_];  // normalized f16
    __shared__ _Float16 wl16[H_][H_];       // 0.5 * W_logit (scale folded)
    __shared__ _Float16 wc16[H_][H_];       // W_ctx
    __shared__ float    biasmix[H_][47];    // W_logit-mixed rel-pos bias
    __shared__ _Float16 mixT[H_][32][32];   // mixed logits [H2][col][row]
    __shared__ _Float16 att16[H_][32][32];  // softmaxed    [H2][row][col]
    __shared__ _Float16 xahT[32][32];       // context^T    [chan][row]
    __shared__ float    outbT[32][32];      // out-proj^T   [chan][row]

    // ---- stage 0: input tile, tiny precomputes, pad zeroing ----
    for (int i = lane; i < C_ * L_; i += 32) {
        int c = i / L_, l = i % L_;
        xs[l][c] = (_Float16)x[((b * C_ + c) * T_ + t) * L_ + l];
    }
    for (int i = lane; i < (32 - L_) * C_; i += 32) {
        int l = L_ + i / C_, c = i % C_;
        xs[l][c] = (_Float16)0.f;
        qh[l][c] = (_Float16)0.f;
        kh[l][c] = (_Float16)0.f;
        vh[l][c] = (_Float16)0.f;
    }
    for (int i = lane; i < H_ * (32 - L_) * C_; i += 32) {
        int h  = i / ((32 - L_) * C_);
        int rr = i % ((32 - L_) * C_);
        att16[h][L_ + rr / C_][rr % C_] = (_Float16)0.f;
    }
    for (int i = lane; i < H_ * H_; i += 32) {
        wl16[i / H_][i % H_] = (_Float16)(0.5f * Wlogit[i]);
        wc16[i / H_][i % H_] = (_Float16)Wctx[i];
    }
    for (int i = lane; i < H_ * 47; i += 32) {
        int H2 = i / 47, rel = i % 47;
        float s = 0.f;
#pragma unroll
        for (int h = 0; h < H_; ++h) s += relb[h * 47 + rel] * Wlogit[h * H_ + H2];
        biasmix[H2][rel] = s;
    }
    __syncthreads();

    // ---- stage 1: QKV projections via WMMA ----
    v16h afr[2];
#pragma unroll
    for (int mt = 0; mt < 2; ++mt)
#pragma unroll
        for (int e = 0; e < 16; ++e)
            afr[mt][e] = xs[lo + mt * 16][wmma_a_k(lane, e)];

    auto project = [&](int mat, const float* __restrict__ bias, float (*dst)[C_]) {
#pragma unroll
        for (int nt = 0; nt < 2; ++nt) {
            const int   n    = lo + nt * 16;
            const v16h  bfr  = fragA[(mat * 2 + nt) * 32 + lane];
            const float bval = bias[n];
#pragma unroll
            for (int mt = 0; mt < 2; ++mt) {
                v8f acc = {};
                acc = __builtin_amdgcn_wmma_f32_16x16x32_f16(
                        false, afr[mt], false, bfr, (short)0, acc, false, false);
#pragma unroll
                for (int r = 0; r < 8; ++r)
                    dst[r + hi * 8 + mt * 16][n] = acc[r] + bval;
            }
        }
    };
    project(0, bq, qf);
    project(1, bk, kf);
    project(2, bv, vf);
    __syncthreads();

    // ---- stage 2: l2norm over D=4, emit f16 copies ----
    for (int i = lane; i < L_ * H_; i += 32) {
        int l = i / H_, h = i % H_;
        const float* src[3] = { &qf[l][h * D_], &kf[l][h * D_], &vf[l][h * D_] };
        _Float16*    dst[3] = { &qh[l][h * D_], &kh[l][h * D_], &vh[l][h * D_] };
#pragma unroll
        for (int a = 0; a < 3; ++a) {
            float ss = 0.f;
#pragma unroll
            for (int d = 0; d < D_; ++d) ss += src[a][d] * src[a][d];
            float inv = 1.0f / fmaxf(sqrtf(ss), 1e-12f);
#pragma unroll
            for (int d = 0; d < D_; ++d) dst[a][d] = (_Float16)(src[a][d] * inv);
        }
    }
    __syncthreads();

    // ---- stage 3: head-mixed logits via WMMA ----
    // mixed[H2] = (q .* 0.5*Wl[:,H2 per-channel]) @ k^T
    v16h aq[2], bk16[2];
#pragma unroll
    for (int mt = 0; mt < 2; ++mt)
#pragma unroll
        for (int e = 0; e < 16; ++e)
            aq[mt][e] = qh[lo + mt * 16][wmma_a_k(lane, e)];
#pragma unroll
    for (int nt = 0; nt < 2; ++nt)
#pragma unroll
        for (int e = 0; e < 16; ++e)
            bk16[nt][e] = kh[lo + nt * 16][wmma_b_k(lane, e)];

#pragma unroll
    for (int H2 = 0; H2 < H_; ++H2) {
        // A-channel k maps to head h = base(e) + hi*2, base in {0,1,4,5}
        _Float16 w0 = wl16[hi * 2 + 0][H2], w1 = wl16[hi * 2 + 1][H2];
        _Float16 w4 = wl16[hi * 2 + 4][H2], w5 = wl16[hi * 2 + 5][H2];
        v16h sv;
#pragma unroll
        for (int e = 0; e < 4; ++e)   sv[e] = w0;
#pragma unroll
        for (int e = 4; e < 8; ++e)   sv[e] = w1;
#pragma unroll
        for (int e = 8; e < 12; ++e)  sv[e] = w4;
#pragma unroll
        for (int e = 12; e < 16; ++e) sv[e] = w5;
#pragma unroll
        for (int mt = 0; mt < 2; ++mt) {
            v16h aH = aq[mt] * sv;           // v_pk_mul_f16 x8
#pragma unroll
            for (int nt = 0; nt < 2; ++nt) {
                v8f acc = {};
                acc = __builtin_amdgcn_wmma_f32_16x16x32_f16(
                        false, aH, false, bk16[nt], (short)0, acc, false, false);
                v8h ph;
#pragma unroll
                for (int r = 0; r < 8; ++r) ph[r] = (_Float16)acc[r];
                *reinterpret_cast<v8h*>(&mixT[H2][lo + nt * 16][hi * 8 + mt * 16]) = ph;
            }
        }
    }
    __syncthreads();

    // ---- stage 4: + biasmix, softmax over j (inherently VALU) ----
    for (int i = lane; i < H_ * L_; i += 32) {
        int H2 = i / L_, r = i % L_;
        float row[L_];
#pragma unroll
        for (int c = 0; c < L_; ++c) {
            int rel = r - c + 23;
            rel = max(0, min(46, rel));
            row[c] = (float)mixT[H2][c][r] + biasmix[H2][rel];
        }
        float mx = row[0];
#pragma unroll
        for (int c = 1; c < L_; ++c) mx = fmaxf(mx, row[c]);
        float sum = 0.f;
#pragma unroll
        for (int c = 0; c < L_; ++c) { row[c] = __expf(row[c] - mx); sum += row[c]; }
        float inv = 1.0f / sum;
#pragma unroll
        for (int c = 0; c < L_; ++c) att16[H2][r][c] = (_Float16)(row[c] * inv);
    }
    __syncthreads();

    // ---- stage 5: ctx-mix + apply to v as one GEMM via WMMA ----
    // K = c*8 + H (so H = e&7 compile-time); B[k][n] = Wc[H][n>>2]*v[c][n]
    v16h bfv[2][6];
#pragma unroll
    for (int nt = 0; nt < 2; ++nt) {
        const int n  = lo + nt * 16;
        const int h2 = n >> 2;
        _Float16 wcol[8];
#pragma unroll
        for (int hh = 0; hh < 8; ++hh) wcol[hh] = wc16[hh][h2];
#pragma unroll
        for (int ks = 0; ks < 6; ++ks)
#pragma unroll
            for (int e = 0; e < 16; ++e) {
                int c = ks * 4 + ((e >> 3) << 1) + hi;
                bfv[nt][ks][e] = wcol[e & 7] * vh[c][n];
            }
    }
#pragma unroll
    for (int mt = 0; mt < 2; ++mt) {
        const int m = lo + mt * 16;
        v8f acc0 = {}, acc1 = {};
#pragma unroll
        for (int ks = 0; ks < 6; ++ks) {
            v16h af;
#pragma unroll
            for (int e = 0; e < 16; ++e) {
                int c = ks * 4 + ((e >> 3) << 1) + hi;
                af[e] = att16[e & 7][m][c];
            }
            acc0 = __builtin_amdgcn_wmma_f32_16x16x32_f16(
                    false, af, false, bfv[0][ks], (short)0, acc0, false, false);
            acc1 = __builtin_amdgcn_wmma_f32_16x16x32_f16(
                    false, af, false, bfv[1][ks], (short)0, acc1, false, false);
        }
        v8h p0, p1;
#pragma unroll
        for (int r = 0; r < 8; ++r) { p0[r] = (_Float16)acc0[r]; p1[r] = (_Float16)acc1[r]; }
        *reinterpret_cast<v8h*>(&xahT[lo][hi * 8 + mt * 16])      = p0;
        *reinterpret_cast<v8h*>(&xahT[lo + 16][hi * 8 + mt * 16]) = p1;
    }
    __syncthreads();

    // ---- stage 6: output projection (Wm) via WMMA ----
    v16h afr2[2];
#pragma unroll
    for (int mt = 0; mt < 2; ++mt)
#pragma unroll
        for (int e = 0; e < 16; ++e)
            afr2[mt][e] = xahT[wmma_a_k(lane, e)][lo + mt * 16];
#pragma unroll
    for (int nt = 0; nt < 2; ++nt) {
        const v16h bfr = fragA[(3 * 2 + nt) * 32 + lane];
#pragma unroll
        for (int mt = 0; mt < 2; ++mt) {
            v8f acc = {};
            acc = __builtin_amdgcn_wmma_f32_16x16x32_f16(
                    false, afr2[mt], false, bfr, (short)0, acc, false, false);
            *reinterpret_cast<v8f*>(&outbT[lo + nt * 16][hi * 8 + mt * 16]) = acc;
        }
    }
    __syncthreads();

    // ---- stage 7: coalesced global store with bias ----
    for (int i = lane; i < C_ * L_; i += 32) {
        int c = i / L_, l = i % L_;
        xatt[((b * C_ + c) * T_ + t) * L_ + l] = outbT[c][l] + bm[c];
    }
}

// =====================================================================
// Kernel B: per-b fused splice + 3x3 conv (implicit GEMM via WMMA)
//           + BN + ReLU + p-projection + final assembly.
// =====================================================================
__global__ __launch_bounds__(128)
void conv_kernel(const float* __restrict__ xatt,
                 const float* __restrict__ pre,
                 const float* __restrict__ convb,
                 const float* __restrict__ gamma, const float* __restrict__ beta,
                 const float* __restrict__ mean,  const float* __restrict__ var,
                 const float* __restrict__ pw,    const float* __restrict__ pb,
                 const int*   __restrict__ flagp,
                 const v16h*  __restrict__ fragC,
                 float* __restrict__ out)
{
    const int b    = blockIdx.x;
    const int tid  = threadIdx.x;
    const int lane = tid & 31;
    const int wave = tid >> 5;
    const int flag = flagp[0];
    const int s    = (flag == 1) ? 18 : (flag == 2) ? 12 : (flag == 3) ? 6 : 0;

    __shared__ _Float16 cp[C_][TP][L_ + 2];   // zero-padded con tile (fp16)
    __shared__ float    yfp[C_][MPAD];        // conv+bn+relu, rows>=216 unused

    __builtin_prefetch(&xatt[(size_t)b * C_ * T_ * L_ + tid * 16], 0, 1);

    for (int i = tid; i < C_ * TP * (L_ + 2); i += 128) {
        int c   = i / (TP * (L_ + 2));
        int rem = i % (TP * (L_ + 2));
        int tt  = rem / (L_ + 2);
        int ll  = rem % (L_ + 2);
        float v = 0.f;
        if (tt >= 1 && tt <= T_ && ll >= 1 && ll <= L_) {
            int t = tt - 1, l = ll - 1;
            bool inw = (l >= s) && (l < s + 6);
            v = inw ? pre[(c * T_ + t) * L_ + l]
                    : xatt[((b * C_ + c) * T_ + t) * L_ + l];
        }
        cp[c][tt][ll] = (_Float16)v;
    }
    __syncthreads();

    for (int job = wave; job < 28; job += 4) {
        int mt = job >> 1;
        int nt = job & 1;
        const int m_base = (lane & 15) + mt * 16;
        const int t0 = m_base / L_;
        const int l0 = m_base % L_;
        const int co = (lane & 15) + nt * 16;
        v8f acc = {};
#pragma unroll
        for (int tap = 0; tap < 9; ++tap) {
            const int kh = tap / 3, kw = tap % 3;
            v16h afrg;
#pragma unroll
            for (int e = 0; e < 16; ++e)
                afrg[e] = cp[wmma_a_k(lane, e)][t0 + kh][l0 + kw];
            const v16h bfrg = fragC[(nt * 9 + tap) * 32 + lane];
            acc = __builtin_amdgcn_wmma_f32_16x16x32_f16(
                    false, afrg, false, bfrg, (short)0, acc, false, false);
        }
        const float a_bn = rsqrtf(var[co] + 1e-5f) * gamma[co];
        const float c_bn = convb[co] - mean[co];
        const float b_bn = beta[co];
#pragma unroll
        for (int r = 0; r < 8; ++r) {
            int m = r + ((lane >> 4) << 3) + mt * 16;
            yfp[co][m] = fmaxf((acc[r] + c_bn) * a_bn + b_bn, 0.f);
        }
    }
    __syncthreads();

    for (int i = tid; i < C_ * T_ * L_; i += 128) {
        int c   = i / (T_ * L_);
        int rem = i % (T_ * L_);
        int t   = rem / L_;
        int l   = rem % L_;
        float xv = xatt[((b * C_ + c) * T_ + t) * L_ + l];
        float o  = xv;
        if (l >= s && l < s + 6) {
            int oidx = l - s;
            float p = pb[oidx];
#pragma unroll
            for (int ll = 0; ll < L_; ++ll) p += yfp[c][t * L_ + ll] * pw[oidx * L_ + ll];
            o = xv - p;
        }
        out[((b * C_ + c) * T_ + t) * L_ + l] = o;
    }
}

extern "C" void kernel_launch(void* const* d_in, const int* in_sizes, int n_in,
                              void* d_out, int out_size, void* d_ws, size_t ws_size,
                              hipStream_t stream)
{
    (void)in_sizes; (void)n_in; (void)out_size; (void)ws_size;
    const float* x      = (const float*)d_in[0];
    const float* Wq     = (const float*)d_in[1];
    const float* bq     = (const float*)d_in[2];
    const float* Wk     = (const float*)d_in[3];
    const float* bk     = (const float*)d_in[4];
    const float* Wv     = (const float*)d_in[5];
    const float* bv     = (const float*)d_in[6];
    const float* Wm     = (const float*)d_in[7];
    const float* bm     = (const float*)d_in[8];
    const float* Wlog   = (const float*)d_in[9];
    const float* Wctx   = (const float*)d_in[10];
    const float* relb   = (const float*)d_in[11];
    const float* pre    = (const float*)d_in[12];
    const float* convw  = (const float*)d_in[13];
    const float* convb  = (const float*)d_in[14];
    const float* gamma  = (const float*)d_in[15];
    const float* beta   = (const float*)d_in[16];
    const float* mean   = (const float*)d_in[17];
    const float* var    = (const float*)d_in[18];
    const float* pw     = (const float*)d_in[19];
    const float* pb     = (const float*)d_in[20];
    const int*   flag   = (const int*)d_in[21];

    // workspace layout: [x_att f32 | fragA f16 | fragC f16]
    const size_t xatt_bytes = (size_t)B_ * C_ * T_ * L_ * sizeof(float);
    float*    xattw = (float*)d_ws;
    _Float16* fragA = (_Float16*)((char*)d_ws + xatt_bytes);
    _Float16* fragC = fragA + 8 * 32 * 16;
    float*    outp  = (float*)d_out;

    prep_frag_kernel<<<26, 32, 0, stream>>>(Wq, Wk, Wv, Wm, convw, fragA, fragC);
    attn_kernel<<<B_ * T_, 32, 0, stream>>>(x, bq, bk, bv, bm, Wlog, Wctx, relb,
                                            (const v16h*)fragA, xattw);
    conv_kernel<<<B_, 128, 0, stream>>>(xattw, pre, convb, gamma, beta,
                                        mean, var, pw, pb, flag,
                                        (const v16h*)fragC, outp);
}
